// Sequence_56796647522962
// MI455X (gfx1250) — matrix operand order repaired and checked by reading, not compiled, not run on hardware
//
#include <hip/hip_runtime.h>
#include <hip/hip_bf16.h>
#include <math.h>

typedef __bf16 bf16_t;
typedef __attribute__((ext_vector_type(16))) __bf16 v16bf;
typedef __attribute__((ext_vector_type(8)))  __bf16 v8bf;
typedef __attribute__((ext_vector_type(8)))  float  v8f;

// Problem dims
#define B_TOT  256
#define T_LEN  2048
#define H_DIM  150
#define G3     450      // 3*H
#define NPAD   512      // 3H padded so 32 N-tiles = exactly 4 per wave (uniform trip count)
#define NTILES 32
#define TPW    4        // tiles per wave
#define KCH    5        // K chunks of 32 (H padded to 160)

// LDS strides
#define GHSTR  513      // f32, odd to spread banks
#define HFSTR  152
#define WSTR   168      // bf16 row stride, 336B = 16B-aligned
#define XCHUNK 128
#define NITER  10       // ceil(150/16) columns per thread in gate phase

// LDS layout: sGh FIRST so WMMA scatter offsets fit the 16-bit DS immediate.
#define OFF_GH   0
#define OFF_Y    (OFF_GH + 16*GHSTR*4)          // 32832
#define OFF_HF   (OFF_Y + 16*17*4)              // 33920
#define OFF_X    (OFF_HF + 16*HFSTR*4)          // 43648
#define OFF_HB   (OFF_X + 16*XCHUNK*4)          // 51840 (16B aligned)
#define OFF_WB   (OFF_HB + 16*WSTR*2)           // 57216 (16B aligned)
#define SMEM_BYTES (OFF_WB + NPAD*WSTR*2)       // 229248 B < 320 KB

__device__ __forceinline__ float fast_sigmoid(float v) {
    return 1.0f / (1.0f + __expf(-v));
}
__device__ __forceinline__ float fast_tanh(float v) {
    return 2.0f * fast_sigmoid(2.0f * v) - 1.0f;
}

__global__ __launch_bounds__(256) void gru_wmma_kernel(
    const float* __restrict__ x,     // [256][2048]
    const float* __restrict__ Wih,   // [450][1]
    const float* __restrict__ Whh,   // [450][150]
    const float* __restrict__ bih,   // [450]
    const float* __restrict__ bhh,   // [450]
    const float* __restrict__ Wlin,  // [1][150]
    const float* __restrict__ blin,  // [1]
    float* __restrict__ out)         // [256][2048]
{
    extern __shared__ char smem[];
    float*  sGh = (float*)(smem + OFF_GH);     // [16][513] raw gh = h@Whh.T
    float*  sY  = (float*)(smem + OFF_Y);      // [16][17]
    float*  sHf = (float*)(smem + OFF_HF);     // [16][152] master h (f32)
    float*  sX  = (float*)(smem + OFF_X);      // [16][128]
    bf16_t* sHb = (bf16_t*)(smem + OFF_HB);    // [16][168] h (bf16, A operand)
    bf16_t* sWb = (bf16_t*)(smem + OFF_WB);    // [512][168] W_hh (bf16, B operand rows)

    const int tid  = threadIdx.x;
    const int wg   = blockIdx.x;           // 16 blocks; batch rows [wg*16, wg*16+16)
    const int lane = tid & 31;
    const int wave = tid >> 5;             // 0..7
    const int mrow = lane & 15;
    const int hi   = lane >> 4;            // lane half (ISA fragment K split)

    // ---------------- Phase 0: stage weights / init state ----------------
    for (int i = tid; i < NPAD*WSTR; i += 256) {
        int j = i / WSTR, k = i - j*WSTR;
        float v = (j < G3 && k < H_DIM) ? Whh[j*H_DIM + k] : 0.0f;
        sWb[i] = (bf16_t)v;
    }
    for (int i = tid; i < 16*WSTR;  i += 256) sHb[i] = (bf16_t)0.0f;
    for (int i = tid; i < 16*HFSTR; i += 256) sHf[i] = 0.0f;

    // Per-thread loop-invariant gate constants (same columns every timestep).
    const int m  = tid & 15;               // batch row within tile
    const int cs = tid >> 4;               // starting column
    float cw_r[NITER], cw_z[NITER], cw_n[NITER];   // W_ih per gate
    float cb_r[NITER], cb_z[NITER];                // combined b_ih+b_hh (r,z)
    float cb_in[NITER], cb_hn[NITER];              // separate biases for n gate
    float cwl[NITER];                              // W_lin
    #pragma unroll
    for (int i = 0; i < NITER; ++i) {
        int c = cs + 16*i;
        bool ok = (c < H_DIM);
        cw_r[i]  = ok ? Wih[c]        : 0.f;
        cw_z[i]  = ok ? Wih[150 + c]  : 0.f;
        cw_n[i]  = ok ? Wih[300 + c]  : 0.f;
        cb_r[i]  = ok ? (bih[c] + bhh[c])             : 0.f;
        cb_z[i]  = ok ? (bih[150 + c] + bhh[150 + c]) : 0.f;
        cb_in[i] = ok ? bih[300 + c] : 0.f;
        cb_hn[i] = ok ? bhh[300 + c] : 0.f;
        cwl[i]   = ok ? Wlin[c] : 0.f;
    }
    const float bl = blin[0];
    __syncthreads();

    // ---------------- Main recurrence over T ----------------
    for (int t = 0; t < T_LEN; ++t) {
        // refill x chunk (consumed only after the post-WMMA barrier)
        if ((t & (XCHUNK - 1)) == 0) {
            for (int i = tid; i < 16*XCHUNK; i += 256) {
                int r = i >> 7, tt = i & (XCHUNK - 1);
                sX[i] = x[(wg*16 + r)*T_LEN + t + tt];
            }
            if (tid < 16 && (t + XCHUNK) < T_LEN)
                __builtin_prefetch(x + (wg*16 + tid)*T_LEN + t + XCHUNK, 0, 1);
        }

        // ---- WMMA phase: gh[16][512] = h[16][160] @ Whh^T (bf16 in, f32 acc)
        v16bf afr[KCH];
        {
            const bf16_t* hrow = sHb + mrow*WSTR;
            #pragma unroll
            for (int kc = 0; kc < KCH; ++kc) {
                int kb = kc*32 + hi*8;
                union { v16bf v; v8bf h[2]; } u;
                u.h[0] = *(const v8bf*)(hrow + kb);
                u.h[1] = *(const v8bf*)(hrow + kb + 16);
                afr[kc] = u.v;
            }
        }
        #pragma unroll
        for (int i4 = 0; i4 < TPW; ++i4) {     // uniform trip count: EXEC stays all-1s
            const int nt = wave*TPW + i4;
            v8f acc = {0.f,0.f,0.f,0.f,0.f,0.f,0.f,0.f};
            const bf16_t* wrowp = sWb + (nt*16 + mrow)*WSTR + hi*8;
            #pragma unroll
            for (int kc = 0; kc < KCH; ++kc) {
                union { v16bf v; v8bf h[2]; } u;
                u.h[0] = *(const v8bf*)(wrowp + kc*32);
                u.h[1] = *(const v8bf*)(wrowp + kc*32 + 16);
                acc = __builtin_amdgcn_wmma_f32_16x16x32_bf16(
                        false, afr[kc], false, u.v, (short)0, acc, false, false);
            }
            // C/D layout: VGPR e, lanes 0-15 -> M=e,N=lane ; lanes 16-31 -> M=e+8,N=lane-16
            // sGh at LDS offset 0 => the 8 row stores fold into DS immediates (e*2052 <= 14364).
            float* g = sGh + (hi*8)*GHSTR + nt*16 + mrow;
            #pragma unroll
            for (int e = 0; e < 8; ++e) g[e*GHSTR] = acc[e];
        }
        __syncthreads();

        // ---- Elementwise gates + state update + head partials
        {
            const float xv = sX[m*XCHUNK + (t & (XCHUNK - 1))];
            const float* ghm = sGh + m*GHSTR;
            float* hfm = sHf + m*HFSTR;
            bf16_t* hbm = sHb + m*WSTR;
            float ypart = 0.0f;
            #pragma unroll
            for (int i = 0; i < NITER; ++i) {
                int c = cs + 16*i;
                if (c < H_DIM) {
                    float r = fast_sigmoid(xv*cw_r[i] + cb_r[i] + ghm[c]);
                    float z = fast_sigmoid(xv*cw_z[i] + cb_z[i] + ghm[150 + c]);
                    float n = fast_tanh   (xv*cw_n[i] + cb_in[i] + r*(ghm[300 + c] + cb_hn[i]));
                    float hp = hfm[c];
                    float hn = fmaxf((1.0f - z)*n + z*hp, 0.0f);
                    hfm[c] = hn;
                    hbm[c] = (bf16_t)hn;
                    ypart += hn * cwl[i];
                }
            }
            sY[m*17 + cs] = ypart;
        }
        __syncthreads();

        // ---- y[b][t] = h_new . W_lin + b_lin  (16-way LDS reduction)
        if (tid < 16) {
            float s = bl;
            #pragma unroll
            for (int i = 0; i < 16; ++i) s += sY[tid*17 + i];
            out[(wg*16 + tid)*T_LEN + t] = s;
        }
    }
}

extern "C" void kernel_launch(void* const* d_in, const int* in_sizes, int n_in,
                              void* d_out, int out_size, void* d_ws, size_t ws_size,
                              hipStream_t stream) {
    (void)in_sizes; (void)n_in; (void)d_ws; (void)ws_size; (void)out_size;
    const float* x    = (const float*)d_in[0];
    const float* Wih  = (const float*)d_in[1];
    const float* Whh  = (const float*)d_in[2];
    const float* bihp = (const float*)d_in[3];
    const float* bhhp = (const float*)d_in[4];
    const float* Wlin = (const float*)d_in[5];
    const float* blin = (const float*)d_in[6];
    float* out = (float*)d_out;

    dim3 grid(B_TOT / 16);   // 16 workgroups, one 16-row batch tile each
    dim3 block(256);         // 8 wave32s
    gru_wmma_kernel<<<grid, block, SMEM_BYTES, stream>>>(
        x, Wih, Whh, bihp, bhhp, Wlin, blin, out);
}
